// AttentionFusionBlock_38268158607374
// MI455X (gfx1250) — matrix operand order repaired and checked by reading, not compiled, not run on hardware
//
#include <hip/hip_runtime.h>
#include <math.h>

// ---------------- CDNA5 WMMA types ----------------
typedef __attribute__((ext_vector_type(16))) __bf16 bf16x16;
typedef __attribute__((ext_vector_type(8)))  __bf16 bf16x8;
typedef __attribute__((ext_vector_type(8)))  float  f32x8;

#define D_      768
#define NHEADS_ 8
#define HD_     96
#define B_      8
#define NT_     256
#define NS_     1024
#define NTOK_   1280
#define LAYERS_ 4
#define ROWS_   (B_*NTOK_)   // 10240

__device__ __forceinline__ float gelu_exact(float x) {
  return 0.5f * x * (1.0f + erff(x * 0.70710678118654752f));
}

// ------------------------------------------------------------------
// Generic batched GEMM:  C[m][c] = sum_k A[m][k] * BT[c][k]  (+bias,+gelu,+resid)
// A: bf16 MxK row-major (lda), BT: bf16 NcolsxK row-major (ldb, K-contiguous)
// One wave computes a (16*MT) x (16*NACC) tile with v_wmma_f32_16x16x32_bf16.
// Explicit ping-pong double buffering (two named register sets, source-level
// unroll by 2): next k-step fragments are loaded before current WMMAs issue,
// so VMEM overlaps matrix math with partial s_wait_loadcnt, and no buffer
// rotation moves are needed. Final-step loads read <=64B past the matrix:
// all GEMM operands live in the workspace, which has a tail guard allocation.
// Fragment layouts per CDNA5 ISA 7.12.2 (wave32):
//   A 16x32 : lane<16 row=lane  K = {0..7,16..23};  lane>=16 row=lane-16 K={8..15,24..31}
//   B 32x16 : lane<16 col=lane  K = {0..7,16..23};  lane>=16 col=lane-16 K={8..15,24..31}
//   C 16x16 : VGPR i -> (row = i + 8*half, col = lane&15)
// All shapes in this pipeline are exact multiples of the tile sizes.
// ------------------------------------------------------------------
template<int MT, int NACC, bool OUT_BF16>
__global__ __launch_bounds__(256) void wmma_gemm(
    const __bf16* __restrict__ A,  int lda, long sA,
    const __bf16* __restrict__ BT, int ldb, long sB,
    void* __restrict__ Cv,         int ldc, long sC,
    const float* __restrict__ bias,
    const float* __restrict__ resid,
    int M, int Ncols, int K, int gelu_flag)
{
  const int batch = blockIdx.y;
  A  += (size_t)batch * sA;
  BT += (size_t)batch * sB;

  const int tilesM = M / (16 * MT);
  const int tilesN = (Ncols + 16 * NACC - 1) / (16 * NACC);
  const int wave   = (int)((blockIdx.x * blockDim.x + threadIdx.x) >> 5);
  if (wave >= tilesM * tilesN) return;            // uniform per wave: EXEC stays full
  const int tm = wave % tilesM;
  const int tn = wave / tilesM;
  const int m0 = tm * 16 * MT;
  const int n0 = tn * 16 * NACC;

  const int lane = threadIdx.x & 31;
  const int half = lane >> 4;
  const int lid  = lane & 15;

  f32x8 acc[MT][NACC] = {};

  const __bf16* arow[MT];
#pragma unroll
  for (int mt = 0; mt < MT; ++mt)
    arow[mt] = A + (size_t)(m0 + mt * 16 + lid) * lda + half * 8;
  const __bf16* brow[NACC];
#pragma unroll
  for (int j = 0; j < NACC; ++j) {
    int c = n0 + j * 16 + lid;
    if (c >= Ncols) c = Ncols - 1;               // clamp (never triggers for our shapes)
    brow[j] = BT + (size_t)c * ldb + half * 8;
  }

  // Two named fragment buffer sets (ping-pong; no rotation copies).
  bf16x8 al0[MT], ah0[MT], bl0[NACC], bh0[NACC];
  bf16x8 al1[MT], ah1[MT], bl1[NACC], bh1[NACC];

#define LOAD_SET(AL, AH, BL, BH, KOFF)                                        \
  do {                                                                        \
    _Pragma("unroll")                                                         \
    for (int mt = 0; mt < MT; ++mt) {                                         \
      AL[mt] = *(const bf16x8*)(arow[mt] + (KOFF));                           \
      AH[mt] = *(const bf16x8*)(arow[mt] + (KOFF) + 16);                      \
    }                                                                         \
    _Pragma("unroll")                                                         \
    for (int j = 0; j < NACC; ++j) {                                          \
      BL[j] = *(const bf16x8*)(brow[j] + (KOFF));                             \
      BH[j] = *(const bf16x8*)(brow[j] + (KOFF) + 16);                        \
    }                                                                         \
  } while (0)

#define MATH_SET(AL, AH, BL, BH)                                              \
  do {                                                                        \
    bf16x16 bfv[NACC];                                                        \
    _Pragma("unroll")                                                         \
    for (int j = 0; j < NACC; ++j)                                            \
      bfv[j] = __builtin_shufflevector(BL[j], BH[j],                          \
                   0,1,2,3,4,5,6,7,8,9,10,11,12,13,14,15);                    \
    _Pragma("unroll")                                                         \
    for (int mt = 0; mt < MT; ++mt) {                                         \
      bf16x16 af = __builtin_shufflevector(AL[mt], AH[mt],                    \
                   0,1,2,3,4,5,6,7,8,9,10,11,12,13,14,15);                    \
      _Pragma("unroll")                                                       \
      for (int j = 0; j < NACC; ++j)                                          \
        acc[mt][j] = __builtin_amdgcn_wmma_f32_16x16x32_bf16(                 \
                         false, af, false, bfv[j], (short)0,                  \
                         acc[mt][j], false, false);                           \
    }                                                                         \
  } while (0)

  LOAD_SET(al0, ah0, bl0, bh0, 0);                 // prologue: k = 0
  const int nsteps = K >> 5;                       // K is a multiple of 32
  int s = 0;
  for (; s + 2 <= nsteps; s += 2) {
    int k1 = (s + 1) << 5;
    __builtin_prefetch(arow[0] + k1 + 224, 0, 1);  // global_prefetch_b8
    LOAD_SET(al1, ah1, bl1, bh1, k1);              // loads in flight...
    MATH_SET(al0, ah0, bl0, bh0);                  // ...while doing step s
    int k2 = (s + 2) << 5;                         // ==K on last pair: guard-padded read
    LOAD_SET(al0, ah0, bl0, bh0, k2);
    MATH_SET(al1, ah1, bl1, bh1);                  // step s+1
  }
  if (s < nsteps)                                  // odd tail (e.g. K=96): math only
    MATH_SET(al0, ah0, bl0, bh0);

#undef LOAD_SET
#undef MATH_SET

  float*  Cf = (float*)Cv;
  __bf16* Cb = (__bf16*)Cv;
#pragma unroll
  for (int mt = 0; mt < MT; ++mt) {
#pragma unroll
    for (int j = 0; j < NACC; ++j) {
      int c = n0 + j * 16 + lid;
      if (c >= Ncols) continue;
#pragma unroll
      for (int i = 0; i < 8; ++i) {
        int m = m0 + mt * 16 + half * 8 + i;
        float v = acc[mt][j][i];
        if (bias)      v += bias[c];
        if (gelu_flag) v  = gelu_exact(v);
        size_t off = (size_t)batch * sC + (size_t)m * ldc + c;
        if (resid)     v += resid[off];
        if (OUT_BF16) Cb[off] = (__bf16)v; else Cf[off] = v;
      }
    }
  }
}

template<int MT, int NACC, bool OUT_BF16>
static void gemm(hipStream_t st,
                 const __bf16* A, int lda, long sA,
                 const __bf16* BT, int ldb, long sB,
                 void* C, int ldc, long sC,
                 const float* bias, const float* resid,
                 int M, int Ncols, int K, int batches, int gelu_flag)
{
  int tiles = (M / (16 * MT)) * ((Ncols + 16 * NACC - 1) / (16 * NACC));
  dim3 grid((tiles + 7) / 8, batches, 1);
  wmma_gemm<MT, NACC, OUT_BF16><<<grid, dim3(256), 0, st>>>(
      A, lda, sA, BT, ldb, sB, C, ldc, sC, bias, resid, M, Ncols, K, gelu_flag);
}

// ---------------- token embed: h = concat(z^T+pos_uav, x^T+pos_sat) ----------------
__global__ __launch_bounds__(256) void embed_kernel(
    const float* __restrict__ z, const float* __restrict__ x,
    const float* __restrict__ pu, const float* __restrict__ ps,
    float* __restrict__ h)
{
  size_t idx = (size_t)blockIdx.x * 256 + threadIdx.x;
  if (idx >= (size_t)ROWS_ * D_) return;
  int    d  = (int)(idx % D_);
  size_t bn = idx / D_;
  int    n  = (int)(bn % NTOK_);
  int    b  = (int)(bn / NTOK_);
  float v;
  if (n < NT_) v = z[((size_t)b * D_ + d) * NT_ + n] + pu[(size_t)n * D_ + d];
  else { int m = n - NT_; v = x[((size_t)b * D_ + d) * NS_ + m] + ps[(size_t)m * D_ + d]; }
  h[idx] = v;
}

// ---------------- weight transpose + f32->bf16 (makes BT K-contiguous) ----------------
__global__ __launch_bounds__(256) void tconv(const float* __restrict__ in,
                                             __bf16* __restrict__ out, int K, int N)
{
  size_t idx = (size_t)blockIdx.x * 256 + threadIdx.x;
  if (idx >= (size_t)K * N) return;
  int n = (int)(idx % N);
  int k = (int)(idx / N);
  out[(size_t)n * K + k] = (__bf16)in[idx];
}

// ---------------- LayerNorm (f32 in, bf16 out), one block per row ----------------
__global__ __launch_bounds__(256) void ln_kernel(const float* __restrict__ X,
                                                 const float* __restrict__ g,
                                                 const float* __restrict__ bt,
                                                 __bf16* __restrict__ Y)
{
  __shared__ float r1[256], r2[256];
  const int t = threadIdx.x;
  const size_t row = blockIdx.x;
  const float* x = X + row * D_;
  float v0 = x[t], v1 = x[t + 256], v2 = x[t + 512];
  r1[t] = v0 + v1 + v2;
  r2[t] = v0 * v0 + v1 * v1 + v2 * v2;
  __syncthreads();
  for (int s = 128; s > 0; s >>= 1) {
    if (t < s) { r1[t] += r1[t + s]; r2[t] += r2[t + s]; }
    __syncthreads();
  }
  float mean = r1[0] * (1.0f / D_);
  float var  = r2[0] * (1.0f / D_) - mean * mean;
  float rstd = rsqrtf(var + 1e-6f);
  __bf16* y = Y + row * D_;
  y[t]       = (__bf16)((v0 - mean) * rstd * g[t]       + bt[t]);
  y[t + 256] = (__bf16)((v1 - mean) * rstd * g[t + 256] + bt[t + 256]);
  y[t + 512] = (__bf16)((v2 - mean) * rstd * g[t + 512] + bt[t + 512]);
}

// ---------------- scatter qkv -> q(b,h,n,e), k(b,h,n,e), v^T(b,h,e,n) ----------------
__global__ __launch_bounds__(256) void repack_qkv(const __bf16* __restrict__ qkv,
    __bf16* __restrict__ q, __bf16* __restrict__ k, __bf16* __restrict__ vt)
{
  size_t idx = (size_t)blockIdx.x * 256 + threadIdx.x;
  if (idx >= (size_t)ROWS_ * D_) return;
  int    d  = (int)(idx % D_);
  size_t bn = idx / D_;
  int    n  = (int)(bn % NTOK_);
  int    b  = (int)(bn / NTOK_);
  int    hh = d / HD_, e = d % HD_;
  const __bf16* src = qkv + bn * (3 * D_);
  size_t base = ((size_t)(b * NHEADS_ + hh) * NTOK_ + n) * HD_ + e;
  q[base] = src[d];
  k[base] = src[D_ + d];
  vt[((size_t)(b * NHEADS_ + hh) * HD_ + e) * NTOK_ + n] = src[2 * D_ + d];
}

// ---------------- row softmax (scaled), f32 scores -> bf16 probs ----------------
__global__ __launch_bounds__(256) void softmax_kernel(const float* __restrict__ S,
                                                      __bf16* __restrict__ P,
                                                      int width, float scale)
{
  __shared__ float red[256];
  const int t = threadIdx.x;
  const size_t row = blockIdx.x;
  const float* s = S + row * (size_t)width;
  __bf16*      p = P + row * (size_t)width;
  float mx = -3.0e38f;
  for (int c = t; c < width; c += 256) mx = fmaxf(mx, s[c] * scale);
  red[t] = mx; __syncthreads();
  for (int st = 128; st > 0; st >>= 1) { if (t < st) red[t] = fmaxf(red[t], red[t + st]); __syncthreads(); }
  mx = red[0]; __syncthreads();
  float ev[5]; float sum = 0.f; int cnt = 0;
  for (int c = t; c < width; c += 256) { float e = expf(s[c] * scale - mx); ev[cnt++] = e; sum += e; }
  red[t] = sum; __syncthreads();
  for (int st = 128; st > 0; st >>= 1) { if (t < st) red[t] += red[t + st]; __syncthreads(); }
  float inv = 1.0f / red[0];
  cnt = 0;
  for (int c = t; c < width; c += 256) p[c] = (__bf16)(ev[cnt++] * inv);
}

// ---------------- head is purely affine: fold w0@w1@w2 into one 768-vector ----------------
__global__ void head_u(const float* __restrict__ w1, const float* __restrict__ w2,
                       float* __restrict__ u) {            // 256 threads
  int j = threadIdx.x;
  float s = 0.f;
  for (int kk = 0; kk < 64; ++kk) s += w1[j * 64 + kk] * w2[kk];
  u[j] = s;
}
__global__ void head_v(const float* __restrict__ w0, const float* __restrict__ u,
                       float* __restrict__ v) {            // 3 x 256 threads
  int i = blockIdx.x * 256 + threadIdx.x;
  if (i >= D_) return;
  float s = 0.f;
  for (int j = 0; j < 256; ++j) s += w0[i * 256 + j] * u[j];
  v[i] = s;
}
__global__ void head_c(const float* __restrict__ w1, const float* __restrict__ w2,
                       const float* __restrict__ b0, const float* __restrict__ b1,
                       const float* __restrict__ b2, float* __restrict__ cout) { // 64 threads
  __shared__ float red[64];
  int kk = threadIdx.x;
  float t = 0.f;
  for (int j = 0; j < 256; ++j) t += b0[j] * w1[j * 64 + kk];
  red[kk] = (t + b1[kk]) * w2[kk];
  __syncthreads();
  for (int st = 32; st > 0; st >>= 1) { if (kk < st) red[kk] += red[kk + st]; __syncthreads(); }
  if (kk == 0) cout[0] = red[0] + b2[0];
}
__global__ __launch_bounds__(256) void head_out(const float* __restrict__ h,
                                                const float* __restrict__ v,
                                                const float* __restrict__ c,
                                                float* __restrict__ out) { // block per sat row
  __shared__ float red[256];
  const int t = threadIdx.x;
  int row = blockIdx.x;                 // 0 .. B*NS-1
  int b = row / NS_, n = row % NS_;
  const float* x = h + ((size_t)b * NTOK_ + NT_ + n) * D_;
  float s = x[t] * v[t] + x[t + 256] * v[t + 256] + x[t + 512] * v[t + 512];
  red[t] = s; __syncthreads();
  for (int st = 128; st > 0; st >>= 1) { if (t < st) red[t] += red[t + st]; __syncthreads(); }
  if (t == 0) out[(size_t)b * NS_ + n] = red[0] + c[0];
}

// ------------------------------------------------------------------
extern "C" void kernel_launch(void* const* d_in, const int* in_sizes, int n_in,
                              void* d_out, int out_size, void* d_ws, size_t ws_size,
                              hipStream_t stream)
{
  const float* z    = (const float*)d_in[0];
  const float* x    = (const float*)d_in[1];
  const float* pu   = (const float*)d_in[2];
  const float* ps   = (const float*)d_in[3];
  const float* ln1g = (const float*)d_in[4];
  const float* ln1b = (const float*)d_in[5];
  const float* qkvw = (const float*)d_in[6];
  const float* qkvb = (const float*)d_in[7];
  const float* projw= (const float*)d_in[8];
  const float* projb= (const float*)d_in[9];
  const float* ln2g = (const float*)d_in[10];
  const float* ln2b = (const float*)d_in[11];
  const float* f1w  = (const float*)d_in[12];
  const float* f1b  = (const float*)d_in[13];
  const float* f2w  = (const float*)d_in[14];
  const float* f2b  = (const float*)d_in[15];
  const float* w0   = (const float*)d_in[16];
  const float* b0   = (const float*)d_in[17];
  const float* w1   = (const float*)d_in[18];
  const float* b1   = (const float*)d_in[19];
  const float* w2   = (const float*)d_in[20];
  const float* b2   = (const float*)d_in[21];
  float* out = (float*)d_out;

  char* ws = (char*)d_ws;
  size_t off = 0;
  auto alloc = [&](size_t bytes) -> char* {
    char* p = ws + off;
    off = (off + bytes + 255) & ~(size_t)255;
    return p;
  };
  float*  h     = (float*) alloc((size_t)ROWS_ * D_ * 4);
  __bf16* ybf   = (__bf16*)alloc((size_t)ROWS_ * D_ * 2);
  __bf16* qkvbf = (__bf16*)alloc((size_t)ROWS_ * 3 * D_ * 2);
  __bf16* q     = (__bf16*)alloc((size_t)B_ * NHEADS_ * NTOK_ * HD_ * 2);
  __bf16* k     = (__bf16*)alloc((size_t)B_ * NHEADS_ * NTOK_ * HD_ * 2);
  __bf16* vt    = (__bf16*)alloc((size_t)B_ * NHEADS_ * HD_ * NTOK_ * 2);
  __bf16* obf   = (__bf16*)alloc((size_t)ROWS_ * D_ * 2);
  __bf16* gbf   = (__bf16*)alloc((size_t)ROWS_ * 4 * D_ * 2);
  float*  Suav  = (float*) alloc((size_t)NHEADS_ * NT_ * NT_ * 4);
  float*  Ssat  = (float*) alloc((size_t)NHEADS_ * NS_ * NTOK_ * 4);
  __bf16* Puav  = (__bf16*)alloc((size_t)NHEADS_ * NT_ * NT_ * 2);
  __bf16* Psat  = (__bf16*)alloc((size_t)NHEADS_ * NS_ * NTOK_ * 2);
  __bf16* qkvT  = (__bf16*)alloc((size_t)LAYERS_ * 3 * D_ * D_ * 2);
  __bf16* projT = (__bf16*)alloc((size_t)LAYERS_ * D_ * D_ * 2);
  __bf16* fc1T  = (__bf16*)alloc((size_t)LAYERS_ * 4 * D_ * D_ * 2);
  __bf16* fc2T  = (__bf16*)alloc((size_t)LAYERS_ * D_ * 4 * D_ * 2);
  float*  u     = (float*) alloc(256 * 4);
  float*  v     = (float*) alloc(D_ * 4);
  float*  cc    = (float*) alloc(4);
  (void)alloc(4096);           // tail guard: pipelined GEMMs over-read <=64B past a matrix
  if (off > ws_size) return;   // workspace too small; bail safely

  // ---- one-time prep (re-done every call: deterministic) ----
  for (int l = 0; l < LAYERS_; ++l) {
    size_t ne;
    ne = (size_t)D_ * 3 * D_;
    tconv<<<(ne + 255) / 256, 256, 0, stream>>>(qkvw + l * ne, qkvT + (size_t)l * 3 * D_ * D_, D_, 3 * D_);
    ne = (size_t)D_ * D_;
    tconv<<<(ne + 255) / 256, 256, 0, stream>>>(projw + l * ne, projT + l * ne, D_, D_);
    ne = (size_t)D_ * 4 * D_;
    tconv<<<(ne + 255) / 256, 256, 0, stream>>>(f1w + l * ne, fc1T + l * ne, D_, 4 * D_);
    ne = (size_t)4 * D_ * D_;
    tconv<<<(ne + 255) / 256, 256, 0, stream>>>(f2w + l * ne, fc2T + l * ne, 4 * D_, D_);
  }
  head_u<<<1, 256, 0, stream>>>(w1, w2, u);
  head_v<<<3, 256, 0, stream>>>(w0, u, v);
  head_c<<<1, 64, 0, stream>>>(w1, w2, b0, b1, b2, cc);

  const size_t tot = (size_t)ROWS_ * D_;
  embed_kernel<<<(tot + 255) / 256, 256, 0, stream>>>(z, x, pu, ps, h);

  const float scale = 0.10206207261596575f;   // 96^-0.5

  for (int l = 0; l < LAYERS_; ++l) {
    ln_kernel<<<ROWS_, 256, 0, stream>>>(h, ln1g + l * D_, ln1b + l * D_, ybf);

    // QKV: (10240x768) @ (768x2304) + b  -> bf16
    gemm<2, 4, true>(stream, ybf, D_, 0, qkvT + (size_t)l * 3 * D_ * D_, D_, 0,
                     qkvbf, 3 * D_, 0, qkvb + l * 3 * D_, nullptr,
                     ROWS_, 3 * D_, D_, 1, 0);
    repack_qkv<<<(tot + 255) / 256, 256, 0, stream>>>(qkvbf, q, k, vt);

    for (int b = 0; b < B_; ++b) {
      const __bf16* qb = q  + (size_t)b * NHEADS_ * NTOK_ * HD_;
      const __bf16* kb = k  + (size_t)b * NHEADS_ * NTOK_ * HD_;
      const __bf16* vb = vt + (size_t)b * NHEADS_ * HD_ * NTOK_;
      // S = Q K^T (batched over 8 heads)
      gemm<2, 4, false>(stream, qb, HD_, (long)NTOK_ * HD_, kb, HD_, (long)NTOK_ * HD_,
                        Suav, NT_, (long)NT_ * NT_, nullptr, nullptr,
                        NT_, NT_, HD_, NHEADS_, 0);
      gemm<2, 4, false>(stream, qb + NT_ * HD_, HD_, (long)NTOK_ * HD_, kb, HD_, (long)NTOK_ * HD_,
                        Ssat, NTOK_, (long)NS_ * NTOK_, nullptr, nullptr,
                        NS_, NTOK_, HD_, NHEADS_, 0);
      softmax_kernel<<<NHEADS_ * NT_, 256, 0, stream>>>(Suav, Puav, NT_, scale);
      softmax_kernel<<<NHEADS_ * NS_, 256, 0, stream>>>(Ssat, Psat, NTOK_, scale);
      // O = P V  (BT = V^T rows are K-contiguous); write heads interleaved into obf
      __bf16* ob_uav = obf + (size_t)b * NTOK_ * D_;
      __bf16* ob_sat = obf + ((size_t)b * NTOK_ + NT_) * D_;
      gemm<2, 2, true>(stream, Puav, NT_, (long)NT_ * NT_, vb, NTOK_, (long)HD_ * NTOK_,
                       ob_uav, D_, (long)HD_, nullptr, nullptr,
                       NT_, HD_, NT_, NHEADS_, 0);
      gemm<2, 2, true>(stream, Psat, NTOK_, (long)NS_ * NTOK_, vb, NTOK_, (long)HD_ * NTOK_,
                       ob_sat, D_, (long)HD_, nullptr, nullptr,
                       NS_, HD_, NTOK_, NHEADS_, 0);
    }

    // h = h + O @ Wp + bp   (in-place residual, f32 out)
    gemm<2, 4, false>(stream, obf, D_, 0, projT + (size_t)l * D_ * D_, D_, 0,
                      h, D_, 0, projb + l * D_, h, ROWS_, D_, D_, 1, 0);

    ln_kernel<<<ROWS_, 256, 0, stream>>>(h, ln2g + l * D_, ln2b + l * D_, ybf);

    // FC1 + exact GELU -> bf16
    gemm<2, 4, true>(stream, ybf, D_, 0, fc1T + (size_t)l * 4 * D_ * D_, D_, 0,
                     gbf, 4 * D_, 0, f1b + l * 4 * D_, nullptr,
                     ROWS_, 4 * D_, D_, 1, 1);
    // h = h + FC2
    gemm<2, 4, false>(stream, gbf, 4 * D_, 0, fc2T + (size_t)l * D_ * 4 * D_, 4 * D_, 0,
                      h, D_, 0, f2b + l * D_, h, ROWS_, D_, 4 * D_, 1, 0);
  }

  head_out<<<B_ * NS_, 256, 0, stream>>>(h, v, cc, out);
}
